// MultiHeadedAttention_17995912970502
// MI455X (gfx1250) — matrix operand order repaired and checked by reading, not compiled, not run on hardware
//
#include <hip/hip_runtime.h>

#define NUM_HEADS 16
#define D_MODEL   1024
#define HEAD_SIZE 64
#define BATCH     2
#define SEQ       2048
#define M_TOK     (BATCH * SEQ)   // 4096

typedef _Float16 half16 __attribute__((ext_vector_type(16)));
typedef _Float16 half8  __attribute__((ext_vector_type(8)));
typedef _Float16 half4  __attribute__((ext_vector_type(4)));
typedef float    float8 __attribute__((ext_vector_type(8)));

// ---------------------------------------------------------------------------
// Fragment helpers (CDNA5 wave32 WMMA 16x16x32 f16 layouts, ISA 7.12.2)
// A-matrix 16x32: lane holds row M = lane%16.
//   lanes 0-15 : VGPR0-3 = K0..7,  VGPR4-7 = K16..23
//   lanes 16-31: VGPR0-3 = K8..15, VGPR4-7 = K24..31
// B-matrix 32x16: lane holds col N = lane%16.
//   lanes 0-15 : VGPR0-3 = K0..7,  VGPR4-7 = K16..23 ; lanes 16-31: VGPR0-3 = K8..15, VGPR4-7 = K24..31
// ---------------------------------------------------------------------------
__device__ __forceinline__ half16 frag_cat(half8 a, half8 b) {
    return __builtin_shufflevector(a, b, 0,1,2,3,4,5,6,7,8,9,10,11,12,13,14,15);
}
__device__ __forceinline__ half16 load_frag_a(const _Float16* rowp, int khalf) {
    const half8 c0 = *(const half8*)(rowp + 8 * khalf);
    const half8 c1 = *(const half8*)(rowp + 16 + 8 * khalf);
    return frag_cat(c0, c1);
}
__device__ __forceinline__ half16 load_frag_b(const _Float16* colp, int khalf) {
    const half8 c0 = *(const half8*)(colp + 8 * khalf);
    const half8 c1 = *(const half8*)(colp + 16 + 8 * khalf);
    return frag_cat(c0, c1);
}
__device__ __forceinline__ float8 wmma32(half16 a, half16 b, float8 c) {
    return __builtin_amdgcn_wmma_f32_16x16x32_f16(false, a, false, b, (short)0, c,
                                                  false, false);
}
__device__ __forceinline__ float half16_max(float x) {
    x = fmaxf(x, __shfl_xor(x, 8, 32));
    x = fmaxf(x, __shfl_xor(x, 4, 32));
    x = fmaxf(x, __shfl_xor(x, 2, 32));
    x = fmaxf(x, __shfl_xor(x, 1, 32));
    return x;
}
__device__ __forceinline__ float half16_sum(float x) {
    x += __shfl_xor(x, 8, 32);
    x += __shfl_xor(x, 4, 32);
    x += __shfl_xor(x, 2, 32);
    x += __shfl_xor(x, 1, 32);
    return x;
}

// ---------------------------------------------------------------------------
// CDNA5 async global->LDS copy (ASYNCcnt path). LDS address = low 32 bits of
// the flat shared pointer (ISA 10.2).
// ---------------------------------------------------------------------------
__device__ __forceinline__ void async_copy_b128(void* lds_ptr, const void* gptr) {
    unsigned loff = (unsigned)(unsigned long long)lds_ptr;
    asm volatile("global_load_async_to_lds_b128 %0, %1, off"
                 :: "v"(loff), "v"(gptr) : "memory");
}
__device__ __forceinline__ void wait_asynccnt0() {
    asm volatile("s_wait_asynccnt 0x0" ::: "memory");
}

// ---------------------------------------------------------------------------
// f32 -> f16 bulk convert
// ---------------------------------------------------------------------------
__global__ __launch_bounds__(256) void cvt_f32_to_f16(const float* __restrict__ in,
                                                      _Float16* __restrict__ out,
                                                      int n4) {
    int i = blockIdx.x * 256 + threadIdx.x;
    if (i < n4) {
        float4 v = ((const float4*)in)[i];
        half4 h = { (_Float16)v.x, (_Float16)v.y, (_Float16)v.z, (_Float16)v.w };
        *(half4*)(out + 4 * (size_t)i) = h;
    }
}

// W [K][N] f32 -> Wt [N][K] f16
__global__ __launch_bounds__(256) void transpose_w_f16(const float* __restrict__ W,
                                                       _Float16* __restrict__ Wt) {
    int t = blockIdx.x * 256 + threadIdx.x;
    int k = t >> 10;
    int n = t & (D_MODEL - 1);
    Wt[(size_t)n * D_MODEL + k] = (_Float16)W[(size_t)k * D_MODEL + n];
}

// ---------------------------------------------------------------------------
// GEMM: C = A[M][1024] x Bt[1024][1024]^T + bias, 64x64 tile per wave.
// mode 0: f16 out, head-grouped  [b,h,s,64]   (Q with scale, K)
// mode 1: f16 out, head-transposed [b,h,64,s] (V)
// mode 2: f32 out, row-major [M][1024]        (final projection)
// ---------------------------------------------------------------------------
__global__ __launch_bounds__(128)
void gemm_wmma_64x64(const _Float16* __restrict__ A,
                     const _Float16* __restrict__ Bt,
                     const float* __restrict__ bias,
                     void* __restrict__ Out, float scale, int mode) {
    const int lane  = threadIdx.x & 31;
    const int nn    = lane & 15;
    const int khalf = lane >> 4;
    const int wid   = (blockIdx.x * 128 + threadIdx.x) >> 5;
    const int tilesN = D_MODEL / 64;                 // 16
    const int tm = (wid / tilesN) * 64;
    const int tn = (wid % tilesN) * 64;

    float8 acc[4][4] = {};
    for (int k0 = 0; k0 < D_MODEL; k0 += 32) {
        half16 af[4], bf[4];
#pragma unroll
        for (int i = 0; i < 4; ++i)
            af[i] = load_frag_a(A + (size_t)(tm + i * 16 + nn) * D_MODEL + k0, khalf);
#pragma unroll
        for (int j = 0; j < 4; ++j)
            bf[j] = load_frag_b(Bt + (size_t)(tn + j * 16 + nn) * D_MODEL + k0, khalf);
#pragma unroll
        for (int i = 0; i < 4; ++i)
#pragma unroll
            for (int j = 0; j < 4; ++j)
                acc[i][j] = wmma32(af[i], bf[j], acc[i][j]);
    }
#pragma unroll
    for (int j = 0; j < 4; ++j) {
        const int col = tn + j * 16 + nn;
        const float bv = bias[col];
#pragma unroll
        for (int i = 0; i < 4; ++i) {
#pragma unroll
            for (int e = 0; e < 8; ++e) {
                const int row = tm + i * 16 + e + 8 * khalf;
                const float v = (acc[i][j][e] + bv) * scale;
                if (mode == 2) {
                    ((float*)Out)[(size_t)row * D_MODEL + col] = v;
                } else {
                    const int b = row >> 11, s = row & (SEQ - 1);
                    const int h = col >> 6,  d = col & 63;
                    size_t idx;
                    if (mode == 0)
                        idx = ((size_t)(b * NUM_HEADS + h) * SEQ + s) * HEAD_SIZE + d;
                    else
                        idx = ((size_t)(b * NUM_HEADS + h) * HEAD_SIZE + d) * SEQ + s;
                    ((_Float16*)Out)[idx] = (_Float16)v;
                }
            }
        }
    }
}

// ---------------------------------------------------------------------------
// Flash attention with cooperative double-buffered async K/V staging.
// 8 waves/block share one (b,h) head; each wave owns a 16-row Q tile.
// K tile [32 keys][64 d] and V tile [64 d][32 keys] staged per block via
// global_load_async_to_lds_b128 (8 KB / 32-key block, once per 8 waves).
// ---------------------------------------------------------------------------
__global__ __launch_bounds__(256)
void attn_flash(const _Float16* __restrict__ Qh, const _Float16* __restrict__ Kh,
                const _Float16* __restrict__ Vt, const unsigned char* __restrict__ maskb,
                _Float16* __restrict__ Ctx) {
    __shared__ __align__(16) _Float16 lds_k[2][32 * 64];   // 2 x 4 KB
    __shared__ __align__(16) _Float16 lds_v[2][64 * 32];   // 2 x 4 KB
    __shared__ __align__(16) _Float16 lds_p[8][16][32];    // 8 KB P staging

    const int lane  = threadIdx.x & 31;
    const int nn    = lane & 15;
    const int khalf = lane >> 4;
    const int wslot = threadIdx.x >> 5;
    const int bh = blockIdx.x >> 4;                 // 32 (b,h) pairs
    const int qt = (blockIdx.x & 15) * 8 + wslot;   // 128 q-tiles per head
    const int b  = bh >> 4;
    const int h  = bh & 15;
    const int q0 = qt * 16;

    const _Float16* Qp = Qh + (size_t)bh * SEQ * HEAD_SIZE;
    const _Float16* Kp = Kh + (size_t)bh * SEQ * HEAD_SIZE;
    const _Float16* Vp = Vt + (size_t)bh * HEAD_SIZE * SEQ;

    // cooperative stage of one 32-key block: 256 threads x (16B K + 16B V)
    const int krow = threadIdx.x >> 3, kch = threadIdx.x & 7;   // 32 rows x 8 chunks
    const int vrow = threadIdx.x >> 2, vch = threadIdx.x & 3;   // 64 rows x 4 chunks
    auto issue_stage = [&](int kb, int buf) {
        async_copy_b128(&lds_k[buf][krow * 64 + kch * 8],
                        Kp + (size_t)(kb + krow) * HEAD_SIZE + kch * 8);
        async_copy_b128(&lds_v[buf][vrow * 32 + vch * 8],
                        Vp + (size_t)vrow * SEQ + kb + vch * 8);
    };

    const half16 qf0 = load_frag_a(Qp + (size_t)(q0 + nn) * HEAD_SIZE, khalf);
    const half16 qf1 = load_frag_a(Qp + (size_t)(q0 + nn) * HEAD_SIZE + 32, khalf);

    float8 o[4] = {};
    float m_i[8], l_i[8];
#pragma unroll
    for (int i = 0; i < 8; ++i) { m_i[i] = -1e30f; l_i[i] = 0.0f; }

    issue_stage(0, 0);
    wait_asynccnt0();
    __syncthreads();

    int buf = 0;
    for (int kb = 0; kb < SEQ; kb += 32) {
        if (kb + 32 < SEQ) issue_stage(kb + 32, buf ^ 1);   // prefetch next block

        const _Float16* kt = lds_k[buf];
        const _Float16* vt = lds_v[buf];
        const half16 k00 = load_frag_b(kt + (size_t)nn * 64, khalf);
        const half16 k01 = load_frag_b(kt + (size_t)nn * 64 + 32, khalf);
        const half16 k10 = load_frag_b(kt + (size_t)(16 + nn) * 64, khalf);
        const half16 k11 = load_frag_b(kt + (size_t)(16 + nn) * 64 + 32, khalf);
        float8 s0 = {}, s1 = {};
        s0 = wmma32(qf0, k00, s0); s0 = wmma32(qf1, k01, s0);
        s1 = wmma32(qf0, k10, s1); s1 = wmma32(qf1, k11, s1);

        const float mk0 = maskb[b * SEQ + kb + nn] ? 0.0f : -1e30f;
        const float mk1 = maskb[b * SEQ + kb + 16 + nn] ? 0.0f : -1e30f;

        float alpha[8];
#pragma unroll
        for (int i = 0; i < 8; ++i) {
            const float a0 = s0[i] + mk0;
            const float a1 = s1[i] + mk1;
            const float rm = half16_max(fmaxf(a0, a1));
            const float mn = fmaxf(m_i[i], rm);
            const float al = __expf(m_i[i] - mn);
            m_i[i] = mn;
            const float p0 = __expf(a0 - mn);
            const float p1 = __expf(a1 - mn);
            l_i[i] = l_i[i] * al + half16_sum(p0 + p1);
            alpha[i] = al;
            const int row = i + 8 * khalf;              // C-layout -> [q][key] in LDS
            lds_p[wslot][row][nn]      = (_Float16)p0;
            lds_p[wslot][row][16 + nn] = (_Float16)p1;
        }
        __builtin_amdgcn_wave_barrier();   // same-wave DS ops are in-order
        const half16 pf = frag_cat(*(const half8*)&lds_p[wslot][nn][8 * khalf],
                                   *(const half8*)&lds_p[wslot][nn][16 + 8 * khalf]);
#pragma unroll
        for (int t = 0; t < 4; ++t) {
#pragma unroll
            for (int e = 0; e < 8; ++e) o[t][e] *= alpha[e];
            const half16 vf = load_frag_b(vt + (size_t)(t * 16 + nn) * 32, khalf);
            o[t] = wmma32(pf, vf, o[t]);
        }

        wait_asynccnt0();   // next buffer landed
        __syncthreads();    // everyone done reading buf; next buf visible to all
        buf ^= 1;
    }
#pragma unroll
    for (int i = 0; i < 8; ++i) {
        const float inv = 1.0f / l_i[i];
        const int row = q0 + i + 8 * khalf;
        const size_t base = ((size_t)(b * SEQ + row)) * D_MODEL + h * HEAD_SIZE;
#pragma unroll
        for (int t = 0; t < 4; ++t)
            Ctx[base + t * 16 + nn] = (_Float16)(o[t][i] * inv);
    }
}

// ---------------------------------------------------------------------------
extern "C" void kernel_launch(void* const* d_in, const int* in_sizes, int n_in,
                              void* d_out, int out_size, void* d_ws, size_t ws_size,
                              hipStream_t stream) {
    (void)in_sizes; (void)n_in; (void)out_size; (void)ws_size;
    const float* k_in = (const float*)d_in[0];
    const float* v_in = (const float*)d_in[1];
    const float* q_in = (const float*)d_in[2];
    const unsigned char* mask = (const unsigned char*)d_in[3];
    const float* Wk = (const float*)d_in[4];  const float* bk = (const float*)d_in[5];
    const float* Wv = (const float*)d_in[6];  const float* bv = (const float*)d_in[7];
    const float* Wq = (const float*)d_in[8];  const float* bq = (const float*)d_in[9];
    const float* Wo = (const float*)d_in[10]; const float* bo = (const float*)d_in[11];
    float* out = (float*)d_out;

    _Float16* ws = (_Float16*)d_ws;
    const size_t TOK = (size_t)M_TOK * D_MODEL;   // 4M elems
    const size_t WSZ = (size_t)D_MODEL * D_MODEL; // 1M elems
    _Float16* k16 = ws;             _Float16* v16 = k16 + TOK;  _Float16* q16 = v16 + TOK;
    _Float16* WkT = q16 + TOK;      _Float16* WvT = WkT + WSZ;
    _Float16* WqT = WvT + WSZ;      _Float16* WoT = WqT + WSZ;
    _Float16* KhW = WoT + WSZ;      _Float16* VtW = KhW + TOK;
    _Float16* QhW = VtW + TOK;      _Float16* CtxW = QhW + TOK;   // total 64 MB

    const int n4 = (int)(TOK / 4);
    cvt_f32_to_f16<<<n4 / 256, 256, 0, stream>>>(k_in, k16, n4);
    cvt_f32_to_f16<<<n4 / 256, 256, 0, stream>>>(v_in, v16, n4);
    cvt_f32_to_f16<<<n4 / 256, 256, 0, stream>>>(q_in, q16, n4);
    transpose_w_f16<<<(int)(WSZ / 256), 256, 0, stream>>>(Wk, WkT);
    transpose_w_f16<<<(int)(WSZ / 256), 256, 0, stream>>>(Wv, WvT);
    transpose_w_f16<<<(int)(WSZ / 256), 256, 0, stream>>>(Wq, WqT);
    transpose_w_f16<<<(int)(WSZ / 256), 256, 0, stream>>>(Wo, WoT);

    gemm_wmma_64x64<<<256, 128, 0, stream>>>(q16, WqT, bq, QhW, 0.125f, 0);
    gemm_wmma_64x64<<<256, 128, 0, stream>>>(k16, WkT, bk, KhW, 1.0f, 0);
    gemm_wmma_64x64<<<256, 128, 0, stream>>>(v16, WvT, bv, VtW, 1.0f, 1);

    attn_flash<<<512, 256, 0, stream>>>(QhW, KhW, VtW, mask, CtxW);

    gemm_wmma_64x64<<<256, 128, 0, stream>>>(CtxW, WoT, bo, out, 1.0f, 2);
}
